// Qwen3CoderNextProjectionLayer_88562225643887
// MI455X (gfx1250) — compile-verified, matrix operand order checked by reading, not run on hardware
//
#include <hip/hip_runtime.h>

// ---------------------------------------------------------------------------
// Qwen3CoderNextProjectionLayer forward for MI455X (gfx1250, wave32, WMMA)
// Heavy GEMMs: v_wmma_f32_16x16x32_bf16, fp32 accumulate, double-buffered LDS
// pipeline; expert-hidden activations kept in bf16; async global->LDS staging
// for bf16 operands via the gfx1250 async builtins.
// ---------------------------------------------------------------------------

typedef __attribute__((ext_vector_type(16))) __bf16 v16bf;
typedef __attribute__((ext_vector_type(8)))  float  v8f;
typedef __attribute__((__vector_size__(16))) int    v4i_t;   // int4 for async-LDS builtin

#define TILE_M 128
#define TILE_N 128
#define TILE_K 32
#define KPAD   8          // LDS row stride = 40 bf16 = 80B: 16B-aligned rows, conflict-free pairs

enum { OP_BIAS = 0, OP_SILU = 1, OP_MOE = 2, OP_BIAS_RESID = 3, OP_SILU_BF16 = 4 };

// Problem constants (from reference)
#define NB   2
#define NS   2048
#define BS   4096      // B*S rows
#define DI   2048      // D_IN == D_INT == D_OUT
#define DC   2560      // D_INT + D_INT/4 (concat width)
#define DE   4096      // 2*D_INT (expert hidden)
#define NE   4         // experts
#define DS4  512       // D_INT/4 (syntax embedding width)

#if defined(__has_builtin)
#if __has_builtin(__builtin_amdgcn_global_load_async_to_lds_b128) && \
    __has_builtin(__builtin_amdgcn_s_wait_asynccnt)
#define HAVE_ASYNC_LDS 1
#endif
#endif
#ifndef HAVE_ASYNC_LDS
#define HAVE_ASYNC_LDS 0
#endif

__device__ __forceinline__ __bf16 f2bf(float f) {
  // gfx1250 has native bf16 converts (v_cvt_*bf16_f32); fptrunc is RNE.
  return (__bf16)f;
}

// ---------------------------------------------------------------------------
// Generic bf16-WMMA GEMM: C[M,N] = epilogue(A[M,K] @ B[K,N])
// Block = 256 threads = 8 waves; block tile 128x128; wave tile 64x32.
// Double-buffered LDS, global loads for tile t+1 overlap WMMAs of tile t.
// ABF16: A operand already bf16 in memory (expert-hidden path).
// ---------------------------------------------------------------------------
template <bool ABF16, int OP>
__global__ __launch_bounds__(256) void gemm_bf16_wmma(
    const void* __restrict__ Ain, int lda,
    const float* __restrict__ B, int ldb,
    void* __restrict__ Cout, int ldc,
    int K,
    const float* __restrict__ bias,
    const float* __restrict__ extra,   // probs (OP_MOE) or residual (OP_BIAS_RESID)
    int expert)
{
  __shared__ __bf16 As[2][TILE_M][TILE_K + KPAD];   // [M][K]
  __shared__ __bf16 Bs[2][TILE_N][TILE_K + KPAD];   // transposed: [N][K]

  const float*  Af = (const float*) Ain;
  const __bf16* Ab = (const __bf16*)Ain;

  const int tid  = threadIdx.x;
  const int lane = tid & 31;
  const int wave = tid >> 5;
  const int lm   = lane & 15;      // M (A frag) / N (B frag, C frag)
  const int hi   = lane >> 4;      // K-half selector for A/B, +8 rows for C

  const int bm = blockIdx.y * TILE_M;
  const int bn = blockIdx.x * TILE_N;
  const int wm = (wave >> 2) * 64;           // 0 / 64
  const int wn = (wave & 3) * 32;            // 0 / 32 / 64 / 96

  v8f zero;
  #pragma unroll
  for (int r = 0; r < 8; ++r) zero[r] = 0.0f;
  v8f acc[4][2];
  #pragma unroll
  for (int i = 0; i < 4; ++i)
    #pragma unroll
    for (int j = 0; j < 2; ++j) acc[i][j] = zero;

  // ---- staging registers -------------------------------------------------
  float4 brf[4];
  float4 arf[4];          // fp32 A path
#if !HAVE_ASYNC_LDS
  uint4  arb[2];          // bf16 A fallback path (32B per thread)
#endif

  auto loadB_g = [&](int k0) {
    #pragma unroll
    for (int i = 0; i < 4; ++i) {
      const int k = i * 8 + (tid >> 5);
      const int n = (tid & 31) * 4;
      brf[i] = *(const float4*)(B + (size_t)(k0 + k) * ldb + bn + n);
    }
  };
  auto storeB_s = [&](int buf) {
    #pragma unroll
    for (int i = 0; i < 4; ++i) {
      const int k = i * 8 + (tid >> 5);
      const int n = (tid & 31) * 4;
      Bs[buf][n + 0][k] = f2bf(brf[i].x);
      Bs[buf][n + 1][k] = f2bf(brf[i].y);
      Bs[buf][n + 2][k] = f2bf(brf[i].z);
      Bs[buf][n + 3][k] = f2bf(brf[i].w);
    }
  };
  auto loadA_g = [&](int k0, int buf) {
    if constexpr (!ABF16) {
      #pragma unroll
      for (int i = 0; i < 4; ++i) {
        const int m = i * 32 + (tid >> 3);
        const int k = (tid & 7) * 4;
        arf[i] = *(const float4*)(Af + (size_t)(bm + m) * lda + k0 + k);
      }
      (void)buf;
    } else {
      const int m = tid >> 1;
      const int k = (tid & 1) * 16;
      const __bf16* g = Ab + (size_t)(bm + m) * lda + k0 + k;
#if HAVE_ASYNC_LDS
      // gfx1250 async copy: global -> LDS directly, tracked by ASYNCcnt.
      // Builtin prototype (from compiler): (int4 AS1*, int4 AS3*, imm, imm).
      __builtin_amdgcn_global_load_async_to_lds_b128(
          (__attribute__((address_space(1))) v4i_t*)g,
          (__attribute__((address_space(3))) v4i_t*)&As[buf][m][k], 0, 0);
      __builtin_amdgcn_global_load_async_to_lds_b128(
          (__attribute__((address_space(1))) v4i_t*)(g + 8),
          (__attribute__((address_space(3))) v4i_t*)&As[buf][m][k + 8], 0, 0);
#else
      arb[0] = *(const uint4*)g;
      arb[1] = *(const uint4*)(g + 8);
      (void)buf;
#endif
    }
  };
  auto storeA_s = [&](int buf) {
    if constexpr (!ABF16) {
      #pragma unroll
      for (int i = 0; i < 4; ++i) {
        const int m = i * 32 + (tid >> 3);
        const int k = (tid & 7) * 4;
        As[buf][m][k + 0] = f2bf(arf[i].x);
        As[buf][m][k + 1] = f2bf(arf[i].y);
        As[buf][m][k + 2] = f2bf(arf[i].z);
        As[buf][m][k + 3] = f2bf(arf[i].w);
      }
    } else {
#if HAVE_ASYNC_LDS
      __builtin_amdgcn_s_wait_asynccnt(0);
      (void)buf;
#else
      const int m = tid >> 1;
      const int k = (tid & 1) * 16;
      *(uint4*)&As[buf][m][k]     = arb[0];
      *(uint4*)&As[buf][m][k + 8] = arb[1];
#endif
    }
  };

  auto compute_tile = [&](int buf) {
    // ISA 16-bit A layout (16x32): VGPR v holds K = {2v | 16+2(v-4)} (+8 if lane>=16)
    v16bf bfrag[2];
    #pragma unroll
    for (int tn = 0; tn < 2; ++tn) {
      const int col = wn + tn * 16 + lm;
      #pragma unroll
      for (int v = 0; v < 8; ++v) {
        const int kk = (v < 4 ? 2 * v : 8 + 2 * v) + 8 * hi;
        bfrag[tn][2 * v]     = Bs[buf][col][kk];
        bfrag[tn][2 * v + 1] = Bs[buf][col][kk + 1];
      }
    }
    #pragma unroll
    for (int tm = 0; tm < 4; ++tm) {
      const int row = wm + tm * 16 + lm;
      v16bf afrag;
      #pragma unroll
      for (int v = 0; v < 8; ++v) {
        const int kk = (v < 4 ? 2 * v : 8 + 2 * v) + 8 * hi;
        afrag[2 * v]     = As[buf][row][kk];
        afrag[2 * v + 1] = As[buf][row][kk + 1];
      }
      #pragma unroll
      for (int tn = 0; tn < 2; ++tn)
        acc[tm][tn] = __builtin_amdgcn_wmma_f32_16x16x32_bf16(
            false, afrag, false, bfrag[tn], (short)0, acc[tm][tn], false, false);
    }
  };

  // ---- pipelined main loop ----------------------------------------------
  const int nk = K / TILE_K;
  loadB_g(0);
  loadA_g(0, 0);
  storeB_s(0);
  storeA_s(0);
  __syncthreads();

  for (int t = 0; t < nk; ++t) {
    const int cur = t & 1, nxt = cur ^ 1;
    const bool has_next = (t + 1 < nk);
    if (has_next) {
      loadB_g((t + 1) * TILE_K);        // in flight during WMMAs below
      loadA_g((t + 1) * TILE_K, nxt);
    }
    if (t + 2 < nk) {
      __builtin_prefetch(Af + (size_t)(bm + (tid >> 3)) * lda + (t + 2) * TILE_K, 0, 1);
      __builtin_prefetch(B + (size_t)((t + 2) * TILE_K + (tid >> 5)) * ldb + bn, 0, 1);
    }
    compute_tile(cur);
    if (has_next) {
      storeB_s(nxt);                    // waits the global loads here
      storeA_s(nxt);
    }
    __syncthreads();
  }

  // ---- epilogue. C/D layout: VGPR r -> M=r+8*hi, lane%16 -> N ------------
  #pragma unroll
  for (int tm = 0; tm < 4; ++tm) {
    #pragma unroll
    for (int tn = 0; tn < 2; ++tn) {
      const int col = bn + wn + tn * 16 + lm;
      const float bv = bias[col];
      #pragma unroll
      for (int r = 0; r < 8; ++r) {
        const int row = bm + wm + tm * 16 + 8 * hi + r;
        const float val = acc[tm][tn][r] + bv;
        const size_t oidx = (size_t)row * ldc + col;
        if constexpr (OP == OP_SILU) {
          ((float*)Cout)[oidx] = val / (1.0f + __expf(-val));      // x*sigmoid(x)
        } else if constexpr (OP == OP_SILU_BF16) {
          ((__bf16*)Cout)[oidx] = f2bf(val / (1.0f + __expf(-val)));
        } else if constexpr (OP == OP_MOE) {
          const float p = extra[(size_t)row * NE + expert];
          const float t2 = p * val;
          float* C = (float*)Cout;
          if (expert == 0) C[oidx] = t2; else C[oidx] += t2;
        } else if constexpr (OP == OP_BIAS_RESID) {
          ((float*)Cout)[oidx] = val + extra[oidx];
        } else {
          ((float*)Cout)[oidx] = val;
        }
      }
    }
  }
}

// ---------------------------------------------------------------------------
// Fill syntax-embedding columns [2048, 2560) of the concat buffer (gather).
// ---------------------------------------------------------------------------
__global__ __launch_bounds__(256) void k_emb(const int* __restrict__ syn,
                                             const float* __restrict__ emb,
                                             float* __restrict__ abuf)
{
  const int row = blockIdx.x;                 // 0..4095
  const int id  = syn[row];
  const float* src = emb + (size_t)id * DS4;
  float* dst = abuf + (size_t)row * DC + DI;
  for (int j = threadIdx.x; j < DS4; j += blockDim.x) dst[j] = src[j];
}

// ---------------------------------------------------------------------------
// Fused: gate = sigmoid(h.W_gate + b); h *= gate; probs = softmax(h.W_r + b_r)
// ---------------------------------------------------------------------------
__global__ __launch_bounds__(256) void k_gate_router(
    float* __restrict__ h,
    const float* __restrict__ Wg, const float* __restrict__ bg,
    const float* __restrict__ Wr, const float* __restrict__ br,
    float* __restrict__ probs)
{
  __shared__ float s[5][256];
  __shared__ float gate_s;
  const int t = threadIdx.x;
  const int row = blockIdx.x;
  float* hrow = h + (size_t)row * DI;

  float pg = 0.f, p0 = 0.f, p1 = 0.f, p2 = 0.f, p3 = 0.f;
  for (int c = t; c < DI; c += 256) {
    const float v = hrow[c];
    pg += v * Wg[c];
    p0 += v * Wr[c * NE + 0]; p1 += v * Wr[c * NE + 1];
    p2 += v * Wr[c * NE + 2]; p3 += v * Wr[c * NE + 3];
  }
  s[0][t] = pg; s[1][t] = p0; s[2][t] = p1; s[3][t] = p2; s[4][t] = p3;
  __syncthreads();
  for (int off = 128; off > 0; off >>= 1) {
    if (t < off)
      #pragma unroll
      for (int q = 0; q < 5; ++q) s[q][t] += s[q][t + off];
    __syncthreads();
  }
  if (t == 0) {
    const float gate = 1.0f / (1.0f + __expf(-(s[0][0] + bg[0])));
    gate_s = gate;
    float l[NE], mx = -1e30f;
    #pragma unroll
    for (int e = 0; e < NE; ++e) { l[e] = gate * s[1 + e][0] + br[e]; mx = fmaxf(mx, l[e]); }
    float sum = 0.f;
    #pragma unroll
    for (int e = 0; e < NE; ++e) { l[e] = __expf(l[e] - mx); sum += l[e]; }
    #pragma unroll
    for (int e = 0; e < NE; ++e) probs[(size_t)row * NE + e] = l[e] / sum;
  }
  __syncthreads();
  const float g = gate_s;
  for (int c = t; c < DI; c += 256) hrow[c] *= g;
}

// h += acc (MoE combine into residual stream)
__global__ __launch_bounds__(256) void k_add(float* __restrict__ h,
                                             const float* __restrict__ a)
{
  const size_t i = (size_t)blockIdx.x * 256 + threadIdx.x;
  h[i] += a[i];
}

// ---------------------------------------------------------------------------
// Grouped Conv1d (groups=256, 8 ch/group, kernel 3, pad 1) + 3-way combine:
// out = h + ctx + conv(h). One thread per (b,s,d) element.
// ---------------------------------------------------------------------------
__global__ __launch_bounds__(256) void k_conv_combine(
    const float* __restrict__ h, const float* __restrict__ ctx,
    const float* __restrict__ cw, const float* __restrict__ cb,
    float* __restrict__ out)
{
  const size_t idx = (size_t)blockIdx.x * 256 + threadIdx.x;  // 0 .. BS*DI-1
  const int d  = (int)(idx & (DI - 1));
  const size_t bs = idx >> 11;                                // b*NS + s
  const int s  = (int)(bs & (NS - 1));
  const size_t brow = bs - s;                                 // b*NS
  const int g  = d >> 3;                                      // group id
  const float* w = cw + (size_t)d * 24;                       // [8][3]

  float sum = cb[d];
  #pragma unroll
  for (int t = 0; t < 3; ++t) {
    const int ss = s + t - 1;
    if (ss >= 0 && ss < NS) {
      const float* hp = h + (brow + ss) * (size_t)DI + g * 8;
      #pragma unroll
      for (int i = 0; i < 8; ++i) sum += w[i * 3 + t] * hp[i];
    }
  }
  out[idx] = h[idx] + ctx[idx] + sum;
}

// ---------------------------------------------------------------------------
// Host-side orchestration (all launches on `stream`, graph-capture safe).
// ---------------------------------------------------------------------------
extern "C" void kernel_launch(void* const* d_in, const int* in_sizes, int n_in,
                              void* d_out, int out_size, void* d_ws, size_t ws_size,
                              hipStream_t stream) {
  (void)in_sizes; (void)n_in; (void)out_size; (void)ws_size;

  const float* x      = (const float*)d_in[0];
  const int*   syn    = (const int*)  d_in[1];
  const float* W_in   = (const float*)d_in[2];
  const float* b_in   = (const float*)d_in[3];
  const float* W_emb  = (const float*)d_in[4];
  const float* W_syn  = (const float*)d_in[5];
  const float* b_syn  = (const float*)d_in[6];
  const float* W_gate = (const float*)d_in[7];
  const float* b_gate = (const float*)d_in[8];
  const float* W_rout = (const float*)d_in[9];
  const float* b_rout = (const float*)d_in[10];
  const float* W_e1   = (const float*)d_in[11];
  const float* b_e1   = (const float*)d_in[12];
  const float* W_e2   = (const float*)d_in[13];
  const float* b_e2   = (const float*)d_in[14];
  const float* W_ctx  = (const float*)d_in[15];
  const float* b_ctx  = (const float*)d_in[16];
  const float* conv_w = (const float*)d_in[17];
  const float* conv_b = (const float*)d_in[18];
  const float* W_out  = (const float*)d_in[19];
  const float* b_out  = (const float*)d_in[20];
  float* out = (float*)d_out;

  // Workspace layout, ~176 MB total (all regions 16B aligned)
  float*  abuf  = (float*)d_ws;                      // [4096][2560] concat / reused as h_final
  float*  hbuf  = abuf  + (size_t)BS * DC;           // [4096][2048]
  float*  accb  = hbuf  + (size_t)BS * DI;           // [4096][2048] MoE accumulator
  float*  ctxb  = accb  + (size_t)BS * DI;           // [4096][2048]
  float*  probs = ctxb  + (size_t)BS * DI;           // [4096][4]
  __bf16* ehbuf = (__bf16*)(probs + (size_t)BS * NE);// [4096][4096] expert hidden (bf16)

  const dim3 blk(256);
  const dim3 gN2048(DI / TILE_N, BS / TILE_M);       // (16, 32)
  const dim3 gN4096(DE / TILE_N, BS / TILE_M);       // (32, 32)

  // 1) h1 = silu(x @ W_in + b_in) -> abuf[:, 0:2048] (ldc = 2560)
  gemm_bf16_wmma<false, OP_SILU><<<gN2048, blk, 0, stream>>>(
      x, DI, W_in, DI, abuf, DC, DI, b_in, nullptr, 0);
  // 2) abuf[:, 2048:2560] = W_syn_emb[syntax_ids]
  k_emb<<<BS, blk, 0, stream>>>(syn, W_emb, abuf);
  // 3) h = concat @ W_syn + b_syn -> hbuf
  gemm_bf16_wmma<false, OP_BIAS><<<gN2048, blk, 0, stream>>>(
      abuf, DC, W_syn, DI, hbuf, DI, DC, b_syn, nullptr, 0);
  // 4) gate + in-place scale + router softmax
  k_gate_router<<<BS, blk, 0, stream>>>(hbuf, W_gate, b_gate, W_rout, b_rout, probs);
  // 5) dense MoE: accb = sum_e probs_e * (silu(h @ W1_e + b1_e) @ W2_e + b2_e)
  for (int e = 0; e < NE; ++e) {
    gemm_bf16_wmma<false, OP_SILU_BF16><<<gN4096, blk, 0, stream>>>(
        hbuf, DI, W_e1 + (size_t)e * DI * DE, DE, ehbuf, DE, DI,
        b_e1 + (size_t)e * DE, nullptr, 0);
    gemm_bf16_wmma<true, OP_MOE><<<gN2048, blk, 0, stream>>>(
        ehbuf, DE, W_e2 + (size_t)e * DE * DI, DI, accb, DI, DE,
        b_e2 + (size_t)e * DI, probs, e);
  }
  // 6) h += moe
  k_add<<<(BS * DI) / 256, blk, 0, stream>>>(hbuf, accb);
  // 7) ctx = h @ W_ctx + b_ctx
  gemm_bf16_wmma<false, OP_BIAS><<<gN2048, blk, 0, stream>>>(
      hbuf, DI, W_ctx, DI, ctxb, DI, DI, b_ctx, nullptr, 0);
  // 8) h_final = h + ctx + groupedConv1d(h)  -> abuf (reused)
  k_conv_combine<<<(BS * DI) / 256, blk, 0, stream>>>(hbuf, ctxb, conv_w, conv_b, abuf);
  // 9) out = h_final @ W_out + b_out + x
  gemm_bf16_wmma<false, OP_BIAS_RESID><<<gN2048, blk, 0, stream>>>(
      abuf, DI, W_out, DI, out, DI, DI, b_out, x, 0);
}
